// GTAN2_14491219657215
// MI455X (gfx1250) — matrix operand order repaired and checked by reading, compiled, not run on hardware
//
#include <hip/hip_runtime.h>
#include <math.h>

#define NN   50000
#define EE   800000
#define NIN  256
#define NHID 128
#define NOUT 64
#define HOPS 10

typedef __attribute__((ext_vector_type(16))) __bf16 v16bf;
typedef __attribute__((ext_vector_type(8)))  __bf16 v8bf;
typedef __attribute__((ext_vector_type(8)))  float  v8f;

// ---------------- elementwise f32 -> bf16 ----------------
__global__ void gtan_cvt_bf16(const float* __restrict__ in, __bf16* __restrict__ out, int n) {
    int i = blockIdx.x * blockDim.x + threadIdx.x;
    if (i < n) out[i] = (__bf16)in[i];
}

// ---------------- WMMA fragment loaders (contiguous k = 16g + s per lane half, same for A and B) ----------------
// A 16x32 bf16: lane m,g (g = lane>>4) holds K = [kb+16g .. kb+16g+15]
__device__ __forceinline__ v16bf gtan_load_fragA(const __bf16* __restrict__ base, int K, int kb, int lane) {
    int m = lane & 15, g = lane >> 4;
    const __bf16* p = base + (size_t)m * K + kb + g * 16;
    v8bf lo = *(const v8bf*)p;
    v8bf hi = *(const v8bf*)(p + 8);
    v16bf r;
#pragma unroll
    for (int i = 0; i < 8; ++i) { r[i] = lo[i]; r[i + 8] = hi[i]; }
    return r;
}
// B 32x16 bf16 (col n of W^T = row n of W): lane n,g holds K = [kb+16g .. kb+16g+15]
__device__ __forceinline__ v16bf gtan_load_fragB(const __bf16* __restrict__ base, int K, int kb, int lane) {
    int n = lane & 15, g = lane >> 4;
    return *(const v16bf*)(base + (size_t)n * K + kb + g * 16);
}

// ---------------- NT GEMM: C[rows,M] = A[rows,K] @ W[M,K]^T + bias ----------------
// One wave per 16x64 output strip (4 col tiles share one A-fragment), 4 waves / block.
// rows % 16 == 0, K % 32 == 0, M % 64 == 0.
__global__ __launch_bounds__(128)
void gtan_wmma_gemm_nt(const __bf16* __restrict__ A, const __bf16* __restrict__ W,
                       const float* __restrict__ bias,
                       float* __restrict__ Cf, __bf16* __restrict__ Cbf,
                       int rows, int K, int M, int doRelu) {
    int lane    = threadIdx.x & 31;
    int waveId  = threadIdx.x >> 5;
    int groupsM = M >> 6;                       // 64-wide column groups
    int strip   = blockIdx.x * 4 + waveId;
    int total   = (rows >> 4) * groupsM;
    if (strip >= total) return;                 // wave-uniform: EXEC stays all-ones
    int rowTile = strip / groupsM;
    int colGrp  = strip - rowTile * groupsM;

    const __bf16* Ab = A + (size_t)rowTile * 16 * K;
    const __bf16* Wb = W + (size_t)colGrp * 64 * K;

    float b0 = 0.f, b1 = 0.f, b2 = 0.f, b3 = 0.f;
    if (bias) {
        const float* bp = bias + colGrp * 64 + (lane & 15);
        b0 = bp[0]; b1 = bp[16]; b2 = bp[32]; b3 = bp[48];
    }
    v8f c0, c1, c2, c3;
#pragma unroll
    for (int r = 0; r < 8; ++r) { c0[r] = b0; c1[r] = b1; c2[r] = b2; c3[r] = b3; }

    for (int kb = 0; kb < K; kb += 32) {
        v16bf af  = gtan_load_fragA(Ab, K, kb, lane);
        v16bf bf0 = gtan_load_fragB(Wb,                     K, kb, lane);
        v16bf bf1 = gtan_load_fragB(Wb + (size_t)16 * K,    K, kb, lane);
        v16bf bf2 = gtan_load_fragB(Wb + (size_t)32 * K,    K, kb, lane);
        v16bf bf3 = gtan_load_fragB(Wb + (size_t)48 * K,    K, kb, lane);
        c0 = __builtin_amdgcn_wmma_f32_16x16x32_bf16(false, af, false, bf0, (short)0, c0, false, false);
        c1 = __builtin_amdgcn_wmma_f32_16x16x32_bf16(false, af, false, bf1, (short)0, c1, false, false);
        c2 = __builtin_amdgcn_wmma_f32_16x16x32_bf16(false, af, false, bf2, (short)0, c2, false, false);
        c3 = __builtin_amdgcn_wmma_f32_16x16x32_bf16(false, af, false, bf3, (short)0, c3, false, false);
    }
    if (doRelu) {
#pragma unroll
        for (int r = 0; r < 8; ++r) {
            c0[r] = fmaxf(c0[r], 0.0f); c1[r] = fmaxf(c1[r], 0.0f);
            c2[r] = fmaxf(c2[r], 0.0f); c3[r] = fmaxf(c3[r], 0.0f);
        }
    }
    int rbase = rowTile * 16 + 8 * (lane >> 4);
    int col   = colGrp * 64 + (lane & 15);
    if (Cf) {
#pragma unroll
        for (int r = 0; r < 8; ++r) {
            float* rp = Cf + (size_t)(rbase + r) * M + col;
            rp[0] = c0[r]; rp[16] = c1[r]; rp[32] = c2[r]; rp[48] = c3[r];
        }
    }
    if (Cbf) {
#pragma unroll
        for (int r = 0; r < 8; ++r) {
            __bf16* rp = Cbf + (size_t)(rbase + r) * M + col;
            rp[0] = (__bf16)c0[r]; rp[16] = (__bf16)c1[r];
            rp[32] = (__bf16)c2[r]; rp[48] = (__bf16)c3[r];
        }
    }
}

// ---------------- per-node attention scalars + seed num/div ----------------
// x1 = xnew.a1 ; x2 = xnew.a2 ; h1 = hW.a2 ; w2 = exp(lrelu(x1+x2))
// div = w2 ; num = w2 * xnew   (edge kernel atomically accumulates on top)
__global__ __launch_bounds__(NHID)
void gtan_attn_node(const float* __restrict__ xnew, const float* __restrict__ hW,
                    const float* __restrict__ a1, const float* __restrict__ a2,
                    float* __restrict__ x1o, float* __restrict__ h1o,
                    float* __restrict__ dvo, float* __restrict__ num) {
    __shared__ float s1[NHID], s2[NHID], s3[NHID];
    __shared__ float w2s;
    int row = blockIdx.x, k = threadIdx.x;
    float xv = xnew[(size_t)row * NHID + k];
    float hv = hW[(size_t)row * NHID + k];
    s1[k] = xv * a1[k];
    s2[k] = xv * a2[k];
    s3[k] = hv * a2[k];
    __syncthreads();
    for (int off = NHID / 2; off > 0; off >>= 1) {
        if (k < off) { s1[k] += s1[k + off]; s2[k] += s2[k + off]; s3[k] += s3[k + off]; }
        __syncthreads();
    }
    if (k == 0) {
        float x1 = s1[0], x2 = s2[0], h1 = s3[0];
        float v  = x1 + x2;
        float w2 = __expf(v > 0.0f ? v : 0.2f * v);
        x1o[row] = x1; h1o[row] = h1; dvo[row] = w2; w2s = w2;
    }
    __syncthreads();
    num[(size_t)row * NHID + k] = w2s * xv;
}

// ---------------- per-edge: w1 = exp(lrelu(x1[s]+h1[t])); scatter-add ----------------
// one wave per edge; lanes stride the 128-wide feature row (4 f32 atomics / lane)
__global__ __launch_bounds__(256)
void gtan_attn_edge(const int* __restrict__ s, const int* __restrict__ t,
                    const float* __restrict__ x1, const float* __restrict__ h1,
                    const float* __restrict__ hW,
                    float* __restrict__ dv, float* __restrict__ num, int E) {
    int wid  = (blockIdx.x * blockDim.x + threadIdx.x) >> 5;
    int lane = threadIdx.x & 31;
    if (wid >= E) return;
    int se = s[wid], te = t[wid];
    float v  = x1[se] + h1[te];
    float w1 = __expf(v > 0.0f ? v : 0.2f * v);
    if (lane == 0) unsafeAtomicAdd(dv + se, w1);
    const float* hr = hW + (size_t)te * NHID;
    float*       nr = num + (size_t)se * NHID;
#pragma unroll
    for (int j = 0; j < 4; ++j) {
        int k = lane + 32 * j;
        unsafeAtomicAdd(nr + k, w1 * hr[k]);
    }
}

// ---------------- h = elu(num/div), emitted as bf16 for the next GEMM ----------------
__global__ __launch_bounds__(256)
void gtan_elu_norm(const float* __restrict__ num, const float* __restrict__ dv,
                   __bf16* __restrict__ hbf, int n) {
    int i = blockIdx.x * blockDim.x + threadIdx.x;
    if (i >= n) return;
    float v = num[i] / dv[i >> 7];
    v = v > 0.0f ? v : (__expf(v) - 1.0f);
    hbf[i] = (__bf16)v;
}

extern "C" void kernel_launch(void* const* d_in, const int* in_sizes, int n_in,
                              void* d_out, int out_size, void* d_ws, size_t ws_size,
                              hipStream_t stream) {
    const float* x    = (const float*)d_in[0];
    const int*   s    = (const int*)d_in[1];
    const int*   t    = (const int*)d_in[2];
    const float* fc1W = (const float*)d_in[3];
    const float* fc1b = (const float*)d_in[4];
    const float* fcsW = (const float*)d_in[5];
    const float* fcsb = (const float*)d_in[6];
    const float* a1   = (const float*)d_in[7];
    const float* a2   = (const float*)d_in[8];
    const float* fc2W = (const float*)d_in[9];
    const float* fc2b = (const float*)d_in[10];
    float* out = (float*)d_out;

    // carve scratch (256B-aligned chunks)
    char* p = (char*)d_ws;
    auto carve = [&](size_t bytes) -> char* {
        char* r = p; p += (bytes + 255) & ~(size_t)255; return r;
    };
    __bf16* x_in_bf = (__bf16*)carve((size_t)NN * NIN * 2);       // bf16 input features
    __bf16* x_bf    = (__bf16*)carve((size_t)NN * NHID * 2);      // fc1 output (constant per hop)
    __bf16* h_bf    = (__bf16*)carve((size_t)NN * NHID * 2);      // hop state
    float*  xnew    = (float*)carve((size_t)NN * NHID * 4);
    float*  hWf     = (float*)carve((size_t)NN * NHID * 4);
    float*  num     = (float*)carve((size_t)NN * NHID * 4);
    float*  x1v     = (float*)carve((size_t)NN * 4);
    float*  h1v     = (float*)carve((size_t)NN * 4);
    float*  dv      = (float*)carve((size_t)NN * 4);
    __bf16* fc1W_bf = (__bf16*)carve((size_t)NHID * NIN * 2);
    __bf16* fcsW_bf = (__bf16*)carve((size_t)HOPS * NHID * NHID * 2);
    __bf16* fc2W_bf = (__bf16*)carve((size_t)NOUT * NHID * 2);

    // ---- convert inputs / weights to bf16 (every call: deterministic, no caching) ----
    {
        int n;
        n = NN * NIN;           gtan_cvt_bf16<<<(n + 255) / 256, 256, 0, stream>>>(x,    x_in_bf, n);
        n = NHID * NIN;         gtan_cvt_bf16<<<(n + 255) / 256, 256, 0, stream>>>(fc1W, fc1W_bf, n);
        n = HOPS * NHID * NHID; gtan_cvt_bf16<<<(n + 255) / 256, 256, 0, stream>>>(fcsW, fcsW_bf, n);
        n = NOUT * NHID;        gtan_cvt_bf16<<<(n + 255) / 256, 256, 0, stream>>>(fc2W, fc2W_bf, n);
    }

    // ---- fc1: x = relu(x @ fc1W^T + b)  -> bf16 only (consumed only by WMMA GEMMs) ----
    {
        int strips = (NN / 16) * (NHID / 64);
        int blocks = (strips + 3) / 4;
        gtan_wmma_gemm_nt<<<blocks, 128, 0, stream>>>(x_in_bf, fc1W_bf, fc1b,
                                                      nullptr, x_bf, NN, NIN, NHID, 1);
    }

    // ---- hops ----
    int stripsHid = (NN / 16) * (NHID / 64);
    int blocksHid = (stripsHid + 3) / 4;
    for (int hop = 0; hop < HOPS; ++hop) {
        const __bf16* Wb  = fcsW_bf + (size_t)hop * NHID * NHID;
        const float*  bb  = fcsb + (size_t)hop * NHID;
        const __bf16* hin = (hop == 0) ? x_bf : h_bf;

        gtan_wmma_gemm_nt<<<blocksHid, 128, 0, stream>>>(x_bf, Wb, bb, xnew, nullptr,
                                                         NN, NHID, NHID, 0);
        gtan_wmma_gemm_nt<<<blocksHid, 128, 0, stream>>>(hin,  Wb, bb, hWf,  nullptr,
                                                         NN, NHID, NHID, 0);
        gtan_attn_node<<<NN, NHID, 0, stream>>>(xnew, hWf,
                                                a1 + (size_t)hop * NHID,
                                                a2 + (size_t)hop * NHID,
                                                x1v, h1v, dv, num);
        gtan_attn_edge<<<(EE + 7) / 8, 256, 0, stream>>>(s, t, x1v, h1v, hWf, dv, num, EE);
        gtan_elu_norm<<<(NN * NHID + 255) / 256, 256, 0, stream>>>(num, dv, h_bf, NN * NHID);
    }

    // ---- fc2: out = h @ fc2W^T + b ----
    {
        int strips = (NN / 16) * (NOUT / 64);
        int blocks = (strips + 3) / 4;
        gtan_wmma_gemm_nt<<<blocks, 128, 0, stream>>>(h_bf, fc2W_bf, fc2b,
                                                      out, nullptr, NN, NHID, NOUT, 0);
    }
}
